// Encoder_62491774156878
// MI455X (gfx1250) — matrix operand order repaired and checked by reading, compiled, not run on hardware
//
#include <hip/hip_runtime.h>
#include <math.h>

// ---------- vector types ----------
typedef __attribute__((ext_vector_type(16))) _Float16 v16h;
typedef __attribute__((ext_vector_type(8)))  _Float16 v8h;
typedef __attribute__((ext_vector_type(8)))  float    v8f;
typedef __attribute__((ext_vector_type(4)))  float    f32x4;
typedef __attribute__((ext_vector_type(4)))  unsigned int u32x4;

union Frag16 { v16h v; u32x4 u[2]; };

// ---------- utility kernels ----------
__global__ void zero_u32(unsigned int* p, int n) {
    int i = blockIdx.x * blockDim.x + threadIdx.x;
    if (i < n) p[i] = 0u;
}

// scale/shift prep: BN folded with conv bias (bnflag=1) or plain bias (bnflag=0)
__global__ void prep_ss(const float* g, const float* be, const float* m, const float* v,
                        const float* b, float* scale, float* shift, int C, int bnflag) {
    int c = blockIdx.x * blockDim.x + threadIdx.x;
    if (c >= C) return;
    if (bnflag) {
        float s = g[c] * rsqrtf(v[c] + 1e-5f);
        scale[c] = s;
        shift[c] = be[c] - m[c] * s + b[c] * s;   // conv bias folded through BN
    } else {
        scale[c] = 1.0f;
        shift[c] = b[c];
    }
}

// weight pack: OIHW f32 -> f16 [O][K], K ordered k = (ky*3+kx)*CI + ci
__global__ void pack_w(const float* __restrict__ w, _Float16* __restrict__ out, int O, int CI) {
    int idx = blockIdx.x * blockDim.x + threadIdx.x;
    int total = O * CI * 9;
    if (idx >= total) return;
    int kx = idx % 3; int r = idx / 3;
    int ky = r % 3;  r /= 3;
    int ci = r % CI; int o = r / CI;
    out[(size_t)o * (9 * CI) + (ky * 3 + kx) * CI + ci] =
        (_Float16)w[(((size_t)o * CI + ci) * 3 + ky) * 3 + kx];
}

// ---------- block 1: direct conv(1->64)+BN+ReLU+pool fused (tiny FLOPs) ----------
__global__ void block1_fused(const float* __restrict__ x, const float* __restrict__ w0,
                             const float* __restrict__ scale, const float* __restrict__ shift,
                             _Float16* __restrict__ out /* [32][64][64][64] NHWC */) {
    int idx = blockIdx.x * blockDim.x + threadIdx.x;
    if (idx >= 32 * 64 * 64 * 64) return;
    int c  = idx & 63; int t = idx >> 6;
    int x2 = t & 63;  t >>= 6;
    int y2 = t & 63;  t >>= 6;
    int n  = t;
    const float* img = x + (size_t)n * 16384;
    float wk[9];
#pragma unroll
    for (int i = 0; i < 9; i++) wk[i] = w0[c * 9 + i];
    float sc = scale[c], sh = shift[c];
    float mx = 0.0f;
#pragma unroll
    for (int py = 0; py < 2; py++)
#pragma unroll
    for (int px = 0; px < 2; px++) {
        int yy0 = y2 * 2 + py, xx0 = x2 * 2 + px;
        float s = 0.0f;
#pragma unroll
        for (int ky = 0; ky < 3; ky++) {
            int yy = yy0 + ky - 1;
            if ((unsigned)yy < 128u) {
#pragma unroll
                for (int kx = 0; kx < 3; kx++) {
                    int xx = xx0 + kx - 1;
                    if ((unsigned)xx < 128u) s += wk[ky * 3 + kx] * img[yy * 128 + xx];
                }
            }
        }
        float v = s * sc + sh;
        v = v > 0.0f ? v : 0.0f;
        mx = v > mx ? v : mx;
    }
    out[idx] = (_Float16)mx;
}

// ---------- 2x2 max pool on NHWC f16 ----------
__global__ void pool2(const _Float16* __restrict__ in, _Float16* __restrict__ out,
                      int Ho, int Wo, int C, int total) {
    int idx = blockIdx.x * blockDim.x + threadIdx.x;
    if (idx >= total) return;
    int c = idx % C; int r = idx / C;
    int xo = r % Wo; r /= Wo;
    int yo = r % Ho; r /= Ho;
    int n = r;
    int Wi = Wo * 2;
    size_t base = (((size_t)n * (Ho * 2) + yo * 2) * Wi + xo * 2) * C + c;
    float a = (float)in[base];
    float b = (float)in[base + C];
    float d = (float)in[base + (size_t)Wi * C];
    float e = (float)in[base + (size_t)Wi * C + C];
    float m = fmaxf(fmaxf(a, b), fmaxf(d, e));
    out[idx] = (_Float16)m;
}

// ---------- WMMA implicit-GEMM 3x3 SAME conv, LDS-staged weights ----------
// Wp: f16 [M][K], K = 9*CI ordered (tap, ci).  X: f16 NHWC, NI images of Hs x Ws x CI.
// Block = 8 waves sharing one 32-channel output tile (bm); per-tap 32xCI weight slab
// staged through LDS, re-read by all 8 waves.  Per wave: 2 otiles x 2 pixel tiles.
// mode 0: out = relu(d*scale+shift) -> f16 NHWC.  mode 1: out = d*scale+shift -> f32 NHWC.
__device__ __forceinline__ void store_tile(v8f acc, int chb, int px, int M, int mode,
                                           const float* scale, const float* shift,
                                           float* outF32, _Float16* outF16) {
    f32x4 s0 = *(const f32x4*)(scale + chb);
    f32x4 s1 = *(const f32x4*)(scale + chb + 4);
    f32x4 h0 = *(const f32x4*)(shift + chb);
    f32x4 h1 = *(const f32x4*)(shift + chb + 4);
    float vout[8];
#pragma unroll
    for (int e = 0; e < 8; e++) {
        float s = (e < 4) ? s0[e] : s1[e - 4];
        float b = (e < 4) ? h0[e] : h1[e - 4];
        float v = acc[e] * s + b;
        if (mode == 0) v = v > 0.0f ? v : 0.0f;
        vout[e] = v;
    }
    if (mode == 0) {
        v8h hv{};
#pragma unroll
        for (int e = 0; e < 8; e++) hv[e] = (_Float16)vout[e];
        *(v8h*)(outF16 + (size_t)px * M + chb) = hv;
    } else {
        f32x4 o0, o1;
#pragma unroll
        for (int e = 0; e < 4; e++) { o0[e] = vout[e]; o1[e] = vout[e + 4]; }
        *(f32x4*)(outF32 + (size_t)px * M + chb)     = o0;
        *(f32x4*)(outF32 + (size_t)px * M + chb + 4) = o1;
    }
}

__global__ void __launch_bounds__(256)
gemm_conv_wmma(const _Float16* __restrict__ Wp, const _Float16* __restrict__ X,
               float* __restrict__ outF32, _Float16* __restrict__ outF16,
               const float* __restrict__ scale, const float* __restrict__ shift,
               int M, int K, int CI, int ciShift, int Hs, int Ws, int Npix, int mode) {
    __shared__ __attribute__((aligned(16))) _Float16 ldsA[32 * 512];   // 32 KB max (CI<=512)

    int nbN = Npix >> 8;                         // 8 waves x 32 pixels per block
    int bm = blockIdx.x / nbN;
    int bn = blockIdx.x - bm * nbN;
    int waveId = threadIdx.x >> 5;
    int lane = threadIdx.x & 31;

    int mrow = lane & 15;
    int kh   = lane >> 4;
    int ncol = lane & 15;
    int HW = Hs * Ws;

    // this wave's two 16-pixel tiles (16 consecutive pixels share one image-row)
    int p0  = (bn * 8 + waveId) * 32;
    int px0 = p0 + ncol;
    int px1 = px0 + 16;
    int n0 = px0 / HW; int r0 = px0 - n0 * HW; int y0 = r0 / Ws; int xx0 = r0 - y0 * Ws;
    int n1 = px1 / HW; int r1 = px1 - n1 * HW; int y1 = r1 / Ws; int xx1 = r1 - y1 * Ws;

    int ciMask8 = (CI >> 3) - 1;                 // chunks of 8 f16 per row
    int rowShift = ciShift - 3;
    int chunks = CI >> 6;                        // 16B-chunks per thread per tap

    v8f acc00 = {}, acc01 = {}, acc10 = {}, acc11 = {};

    for (int tap = 0; tap < 9; ++tap) {
        int dy3 = tap / 3;
        int dy = dy3 - 1;
        int dx = tap - dy3 * 3 - 1;

        // ---- cooperative stage: 32 x CI weight slab for this tap into LDS ----
        __syncthreads();
        {
            const _Float16* wsrc = Wp + (size_t)bm * 32 * K + (size_t)tap * CI;
            for (int j = 0; j < chunks; ++j) {
                int cid = threadIdx.x + (j << 8);
                int row = cid >> rowShift;
                int col = cid & ciMask8;
                *(u32x4*)(&ldsA[row * CI + (col << 3)]) =
                    *(const u32x4*)(wsrc + (size_t)row * K + (col << 3));
            }
            if (tap + 1 < 9)
                __builtin_prefetch(wsrc + CI + (size_t)threadIdx.x * 8, 0, 3);  // next slab
        }
        __syncthreads();

        // ---- per-tap B base pointers + border predicates (hoisted) ----
        int yA = y0 + dy, xA = xx0 + dx;
        int yB = y1 + dy, xB = xx1 + dx;
        bool vA = (unsigned)yA < (unsigned)Hs && (unsigned)xA < (unsigned)Ws;
        bool vB = (unsigned)yB < (unsigned)Hs && (unsigned)xB < (unsigned)Ws;
        const _Float16* xpA = X + (((size_t)n0 * Hs + yA) * Ws + xA) * CI;
        const _Float16* xpB = X + (((size_t)n1 * Hs + yB) * Ws + xB) * CI;

        const _Float16* la0 = &ldsA[mrow * CI + kh * 8];
        const _Float16* la1 = &ldsA[(mrow + 16) * CI + kh * 8];

        // ---- hot loop: pure load + wmma ----
        for (int ci = 0; ci < CI; ci += 32) {
            Frag16 a0, a1;
            a0.u[0] = *(const u32x4*)(la0 + ci);
            a0.u[1] = *(const u32x4*)(la0 + ci + 16);
            a1.u[0] = *(const u32x4*)(la1 + ci);
            a1.u[1] = *(const u32x4*)(la1 + ci + 16);

            Frag16 b0f, b1f;
            int cb = ci + kh * 8;
            if (vA) {
                b0f.u[0] = *(const u32x4*)(xpA + cb);
                b0f.u[1] = *(const u32x4*)(xpA + cb + 16);
            } else { b0f.u[0] = u32x4{}; b0f.u[1] = u32x4{}; }
            if (vB) {
                b1f.u[0] = *(const u32x4*)(xpB + cb);
                b1f.u[1] = *(const u32x4*)(xpB + cb + 16);
            } else { b1f.u[0] = u32x4{}; b1f.u[1] = u32x4{}; }

            acc00 = __builtin_amdgcn_wmma_f32_16x16x32_f16(false, a0.v, false, b0f.v, (short)0, acc00, false, false);
            acc01 = __builtin_amdgcn_wmma_f32_16x16x32_f16(false, a0.v, false, b1f.v, (short)0, acc01, false, false);
            acc10 = __builtin_amdgcn_wmma_f32_16x16x32_f16(false, a1.v, false, b0f.v, (short)0, acc10, false, false);
            acc11 = __builtin_amdgcn_wmma_f32_16x16x32_f16(false, a1.v, false, b1f.v, (short)0, acc11, false, false);
        }
    }

    // ---- epilogue: each lane owns 8 contiguous channels of one pixel ----
    int chb0 = bm * 32 + kh * 8;        // otile 0
    int chb1 = chb0 + 16;               // otile 1
    store_tile(acc00, chb0, px0, M, mode, scale, shift, outF32, outF16);
    store_tile(acc01, chb0, px1, M, mode, scale, shift, outF32, outF16);
    store_tile(acc10, chb1, px0, M, mode, scale, shift, outF32, outF16);
    store_tile(acc11, chb1, px1, M, mode, scale, shift, outF32, outF16);
}

// ---------- concat [x | h] -> XH (NHWC f16, 512 ch) ----------
__global__ void concat_xh(const _Float16* __restrict__ xsrc, int t, int T,
                          const _Float16* __restrict__ hsrc, _Float16* __restrict__ xh) {
    int idx = blockIdx.x * blockDim.x + threadIdx.x;
    if (idx >= 2 * 16 * 16 * 512) return;
    int c = idx & 511;
    int p = idx >> 9;              // pixel (b,y,x), b = p>>8, yx = p&255
    int b = p >> 8;
    int yx = p & 255;
    if (c < 256)
        xh[idx] = xsrc[((size_t)(b * T + t) * 256 + yx) * 256 + c];
    else
        xh[idx] = hsrc[(size_t)p * 256 + (c - 256)];
}

// ---------- fused LSTM gate math ----------
__global__ void lstm_gates(const float* __restrict__ z, float* __restrict__ c_st,
                           _Float16* __restrict__ h16, float* __restrict__ zseq,
                           float* __restrict__ zlast, int t, int writeOut) {
    int idx = blockIdx.x * blockDim.x + threadIdx.x;
    if (idx >= 2 * 256 * 256) return;
    int hid = idx & 255;
    int p = idx >> 8;
    size_t base = (size_t)p * 1024;
    float zi = z[base + hid];
    float zf = z[base + 256 + hid];
    float zo = z[base + 512 + hid];
    float zg = z[base + 768 + hid];
    float si = 1.0f / (1.0f + expf(-zi));
    float sf = 1.0f / (1.0f + expf(-zf));
    float so = 1.0f / (1.0f + expf(-zo));
    float cn = sf * c_st[idx] + si * tanhf(zg);
    float h = so * tanhf(cn);
    c_st[idx] = cn;
    h16[idx] = (_Float16)h;
    if (writeOut) {
        int b = p >> 8;
        int yx = p & 255;
        zseq[(((size_t)b * 16 + t) * 256 + hid) * 256 + yx] = h;   // (B,T,256,16,16)
        if (t == 15)
            zlast[((size_t)b * 256 + hid) * 256 + yx] = h;         // (B,256,16,16)
    }
}

// ---------- host orchestration ----------
extern "C" void kernel_launch(void* const* d_in, const int* in_sizes, int n_in,
                              void* d_out, int out_size, void* d_ws, size_t ws_size,
                              hipStream_t stream) {
    (void)in_sizes; (void)n_in; (void)out_size; (void)ws_size;
    const float* x   = (const float*)d_in[0];
    const float* w0  = (const float*)d_in[1];
    const float* b0  = (const float*)d_in[2];
    const float* g0  = (const float*)d_in[3];
    const float* be0 = (const float*)d_in[4];
    const float* m0  = (const float*)d_in[5];
    const float* v0  = (const float*)d_in[6];
    const float* w1  = (const float*)d_in[7];
    const float* b1  = (const float*)d_in[8];
    const float* g1  = (const float*)d_in[9];
    const float* be1 = (const float*)d_in[10];
    const float* m1  = (const float*)d_in[11];
    const float* v1  = (const float*)d_in[12];
    const float* w2  = (const float*)d_in[13];
    const float* b2  = (const float*)d_in[14];
    const float* g2  = (const float*)d_in[15];
    const float* be2 = (const float*)d_in[16];
    const float* m2  = (const float*)d_in[17];
    const float* v2  = (const float*)d_in[18];
    const float* lw0 = (const float*)d_in[19];
    const float* lb0 = (const float*)d_in[20];
    const float* lw1 = (const float*)d_in[21];
    const float* lb1 = (const float*)d_in[22];

    float* zseq  = (float*)d_out;
    float* zlast = zseq + (size_t)2 * 16 * 256 * 256;

    char* base = (char*)d_ws;
    size_t off = 0;
    auto alloc = [&](size_t bytes) -> void* {
        void* p = base + off;
        off += (bytes + 255) & ~(size_t)255;
        return p;
    };

    _Float16* w1p  = (_Float16*)alloc((size_t)128 * 576 * 2);
    _Float16* w2p  = (_Float16*)alloc((size_t)256 * 1152 * 2);
    _Float16* lw0p = (_Float16*)alloc((size_t)1024 * 4608 * 2);
    _Float16* lw1p = (_Float16*)alloc((size_t)1024 * 4608 * 2);
    float* sc0 = (float*)alloc(64 * 4);    float* sh0 = (float*)alloc(64 * 4);
    float* sc1 = (float*)alloc(128 * 4);   float* sh1 = (float*)alloc(128 * 4);
    float* sc2 = (float*)alloc(256 * 4);   float* sh2 = (float*)alloc(256 * 4);
    float* scL0 = (float*)alloc(1024 * 4); float* shL0 = (float*)alloc(1024 * 4);
    float* scL1 = (float*)alloc(1024 * 4); float* shL1 = (float*)alloc(1024 * 4);
    _Float16* s0 = (_Float16*)alloc((size_t)32 * 64 * 64 * 64 * 2);
    _Float16* s1 = (_Float16*)alloc((size_t)32 * 64 * 64 * 128 * 2);
    _Float16* s2 = (_Float16*)alloc((size_t)32 * 32 * 32 * 128 * 2);
    _Float16* s3 = (_Float16*)alloc((size_t)32 * 32 * 32 * 256 * 2);
    _Float16* s4 = (_Float16*)alloc((size_t)32 * 16 * 16 * 256 * 2);
    _Float16* xh = (_Float16*)alloc((size_t)2 * 16 * 16 * 512 * 2);
    float* zbuf  = (float*)alloc((size_t)2 * 16 * 16 * 1024 * 4);
    _Float16* h1 = (_Float16*)alloc((size_t)131072 * 2);
    _Float16* h2 = (_Float16*)alloc((size_t)131072 * 2);
    float* c1    = (float*)alloc((size_t)131072 * 4);
    float* c2    = (float*)alloc((size_t)131072 * 4);

    // init recurrent state
    zero_u32<<<512, 256, 0, stream>>>((unsigned int*)c1, 131072);
    zero_u32<<<512, 256, 0, stream>>>((unsigned int*)c2, 131072);
    zero_u32<<<256, 256, 0, stream>>>((unsigned int*)h1, 65536);
    zero_u32<<<256, 256, 0, stream>>>((unsigned int*)h2, 65536);

    // BN scale/shift (conv bias folded) + LSTM bias
    prep_ss<<<1, 256, 0, stream>>>(g0, be0, m0, v0, b0, sc0, sh0, 64, 1);
    prep_ss<<<1, 256, 0, stream>>>(g1, be1, m1, v1, b1, sc1, sh1, 128, 1);
    prep_ss<<<1, 256, 0, stream>>>(g2, be2, m2, v2, b2, sc2, sh2, 256, 1);
    prep_ss<<<4, 256, 0, stream>>>(nullptr, nullptr, nullptr, nullptr, lb0, scL0, shL0, 1024, 0);
    prep_ss<<<4, 256, 0, stream>>>(nullptr, nullptr, nullptr, nullptr, lb1, scL1, shL1, 1024, 0);

    // pack weights to f16 [O][K]
    pack_w<<<(73728 + 255) / 256, 256, 0, stream>>>(w1, w1p, 128, 64);
    pack_w<<<(294912 + 255) / 256, 256, 0, stream>>>(w2, w2p, 256, 128);
    pack_w<<<(4718592 + 255) / 256, 256, 0, stream>>>(lw0, lw0p, 1024, 512);
    pack_w<<<(4718592 + 255) / 256, 256, 0, stream>>>(lw1, lw1p, 1024, 512);

    // CNN block 1 (direct, fused pool) -> s0: [32][64][64][64]
    block1_fused<<<32768, 256, 0, stream>>>(x, w0, sc0, sh0, s0);

    // CNN block 2: WMMA conv 64->128 @64x64 -> s1, pool -> s2
    // grid = (M/32) * (Npix/256)
    gemm_conv_wmma<<<4 * 512, 256, 0, stream>>>(w1p, s0, nullptr, s1, sc1, sh1,
                                                128, 576, 64, 6, 64, 64, 32 * 64 * 64, 0);
    pool2<<<16384, 256, 0, stream>>>(s1, s2, 32, 32, 128, 32 * 32 * 32 * 128);

    // CNN block 3: WMMA conv 128->256 @32x32 -> s3, pool -> s4
    gemm_conv_wmma<<<8 * 128, 256, 0, stream>>>(w2p, s2, nullptr, s3, sc2, sh2,
                                                256, 1152, 128, 7, 32, 32, 32 * 32 * 32, 0);
    pool2<<<8192, 256, 0, stream>>>(s3, s4, 16, 16, 256, 32 * 16 * 16 * 256);

    // ConvLSTM: 16 sequential steps, 2 layers, WMMA GEMM M=1024 K=4608 N=512
    for (int t = 0; t < 16; t++) {
        concat_xh<<<1024, 256, 0, stream>>>(s4, t, 16, h1, xh);
        gemm_conv_wmma<<<32 * 2, 256, 0, stream>>>(lw0p, xh, zbuf, nullptr, scL0, shL0,
                                                   1024, 4608, 512, 9, 16, 16, 512, 1);
        lstm_gates<<<512, 256, 0, stream>>>(zbuf, c1, h1, nullptr, nullptr, t, 0);

        concat_xh<<<1024, 256, 0, stream>>>(h1, 0, 1, h2, xh);
        gemm_conv_wmma<<<32 * 2, 256, 0, stream>>>(lw1p, xh, zbuf, nullptr, scL1, shL1,
                                                   1024, 4608, 512, 9, 16, 16, 512, 1);
        lstm_gates<<<512, 256, 0, stream>>>(zbuf, c2, h2, zseq, zlast, t, 1);
    }
}
